// QuantLinear_19456201851575
// MI455X (gfx1250) — compile-verified, matrix operand order checked by reading, not run on hardware
//
#include <hip/hip_runtime.h>
#include <hip/hip_bf16.h>

typedef __attribute__((ext_vector_type(16))) _Float16 v16h;
typedef __attribute__((ext_vector_type(8)))  _Float16 v8h;
typedef __attribute__((ext_vector_type(2)))  _Float16 h2;
typedef __attribute__((ext_vector_type(8)))  float    v8f;
typedef __attribute__((ext_vector_type(4)))  float    v4f;

#define BLOCK_M 256
#define BLOCK_N 128
#define BLOCK_K 32
#define GROUPSZ 128
#define LDS_STRIDE (BLOCK_K + 8)   // pad rows to dodge bank conflicts

// W4A16 GEMM: out[M,N] = x[M,K] @ dequant(qweight)[K,N] + bias
//   - int4 dequantized on the fly entirely in the f16 domain:
//       (0x6400|q) - (0x6400|(z+1)) is exact (integers 1024..1040 in f16),
//       then one v_pk_mul_f16 by the f16 scale. ~15 VALU per 8 nibbles.
//   - nibble-pair extraction (n_j, n_{j+4}) permutes K within 8-wide groups;
//     A is staged with the identical permutation (free: just pairs the f32->f16
//     packing differently), so the WMMA K-reduction is unchanged.
//   - v_wmma_f32_16x16x32_f16, f32 accumulation
//   - 256 threads = 8 wave32; block tile 256x128; wave tile 64x64 (4x4 WMMA)
//   - ping-pong LDS double buffering: one barrier per K-step; global loads for
//     step k+1 are in flight during step k's 16-WMMA burst
__global__ __launch_bounds__(256)
void w4a16_gemm_wmma(const float* __restrict__ x,
                     const int*   __restrict__ qweight,   // [K/8, N] 8 nibbles along K
                     const int*   __restrict__ qzeros,    // [K/128, N/8] 8 nibbles along N
                     const float* __restrict__ scales,    // [K/128, N]
                     const float* __restrict__ bias,      // [N]
                     float* __restrict__ out,
                     int M, int N, int K)
{
    __shared__ alignas(16) _Float16 sA[2][BLOCK_M][LDS_STRIDE]; // activations (f16)
    __shared__ alignas(16) _Float16 sB[2][BLOCK_N][LDS_STRIDE]; // dequant weights, N-major

    const int tid  = threadIdx.x;
    const int lane = tid & 31;
    const int wave = tid >> 5;

    const int m0 = blockIdx.y * BLOCK_M;
    const int n0 = blockIdx.x * BLOCK_N;

    // 8 waves -> 4 (M) x 2 (N); each wave owns a 64x64 sub-tile = 4x4 WMMA tiles
    const int wm = (wave & 3) * 64;    // 0,64,128,192
    const int wn = (wave >> 2) * 64;   // 0,64

    v8f acc[4][4];
    {
        const v8f vzero = {0.f,0.f,0.f,0.f,0.f,0.f,0.f,0.f};
        #pragma unroll
        for (int i = 0; i < 4; ++i)
            #pragma unroll
            for (int j = 0; j < 4; ++j)
                acc[i][j] = vzero;
    }

    // A staging: thread t owns one full row slice: row = t, cols k0..k0+31
    const int a_row = tid;                              // 0..255
    const float* aPtr = x + (size_t)(m0 + a_row) * K;   // + k0 at use

    // B staging: 512 packed int32 per K-step (4 rows x 128 cols); 2 per thread,
    // both at the same output column -> one zero/scale fetch per thread.
    const int n_loc   = tid & 127;     // 0..127 (column within tile)
    const int kr0     = tid >> 7;      // 0 or 1 -> packed rows {kr0, kr0+2}
    const int n_glb   = n0 + n_loc;
    const int z_shift = (n_glb & 7) * 4;

    const int l16    = lane & 15;
    const int half16 = (lane >> 4) * 16;   // K sub-slice selector for fragments

    // ---- staging registers (issued one step ahead) ----
    v4f      aReg[8];
    unsigned wReg[2];
    h2       zh2, sh2;   // f16-domain zero bias (1024+z+1 replicated) and scale

    // issue global loads for K-step at k0 into registers
    auto issue_stage = [&](int k0) {
        #pragma unroll
        for (int i = 0; i < 8; ++i)
            aReg[i] = *(const v4f*)(aPtr + k0 + i * 4);
        const int g  = k0 / GROUPSZ;
        const int zp = qzeros[(size_t)g * (N >> 3) + (n_glb >> 3)];
        const unsigned zq = ((unsigned)(zp >> z_shift) & 15u) + 1u;   // stored as zero-1
        zh2 = __builtin_bit_cast(h2, 0x64006400u | zq | (zq << 16));  // {1024+zq, 1024+zq}
        const _Float16 sh = (_Float16)scales[(size_t)g * N + n_glb];
        sh2[0] = sh; sh2[1] = sh;
        #pragma unroll
        for (int h = 0; h < 2; ++h)
            wReg[h] = (unsigned)qweight[(size_t)((k0 >> 3) + kr0 + h * 2) * N + n_glb];
    };

    // convert/dequant staged registers and store into LDS buffer `buf`
    auto commit_stage = [&](int buf) {
        // A: f32 -> f16 with the (t, t+4) K-pair permutation
        #pragma unroll
        for (int i = 0; i < 4; ++i) {
            v8h h;
            #pragma unroll
            for (int t = 0; t < 4; ++t) {
                h[2 * t]     = (_Float16)aReg[2 * i][t];       // K = 8i + t
                h[2 * t + 1] = (_Float16)aReg[2 * i + 1][t];   // K = 8i + t + 4
            }
            *(v8h*)&sA[buf][a_row][i * 8] = h;
        }
        // B: f16-domain dequant, pairs (n_j, n_{j+4}) -> same K permutation
        #pragma unroll
        for (int h = 0; h < 2; ++h) {
            const int      kr = kr0 + h * 2;          // 0..3 (8 K-values each)
            const unsigned w  = wReg[h];
            v8h hw;
            #pragma unroll
            for (int j = 0; j < 4; ++j) {
                const unsigned bits = ((w >> (4 * j)) & 0x000F000Fu) | 0x64006400u;
                const h2 q  = __builtin_bit_cast(h2, bits);    // {1024+n_j, 1024+n_{j+4}}
                const h2 d  = (q - zh2) * sh2;                 // exact sub, one rounding
                hw[2 * j]     = d[0];
                hw[2 * j + 1] = d[1];
            }
            *(v8h*)&sB[buf][n_loc][kr * 8] = hw;
        }
    };

    // ---- prologue: fill buffer 0 ----
    issue_stage(0);
    commit_stage(0);
    __syncthreads();

    // ---- main pipeline: one barrier per 32-K step ----
    for (int k0 = 0; k0 < K; k0 += BLOCK_K) {
        const int  p       = (k0 >> 5) & 1;
        const bool hasNext = (k0 + BLOCK_K) < K;

        if (hasNext) {
            issue_stage(k0 + BLOCK_K);           // global loads in flight during WMMAs
            if (k0 + 2 * BLOCK_K < K) {          // prefetch two steps out
                __builtin_prefetch(aPtr + k0 + 2 * BLOCK_K, 0, 3);
                __builtin_prefetch(qweight + (size_t)(((k0 + 2 * BLOCK_K) >> 3) + kr0) * N + n_glb, 0, 3);
            }
        }

        // 16-WMMA burst from buffer p
        v16h bfrag[4];
        #pragma unroll
        for (int ni = 0; ni < 4; ++ni) {
            union { v16h v; v8h h[2]; } u;
            u.h[0] = *(const v8h*)&sB[p][wn + ni * 16 + l16][half16];
            u.h[1] = *(const v8h*)&sB[p][wn + ni * 16 + l16][half16 + 8];
            bfrag[ni] = u.v;
        }
        #pragma unroll
        for (int mi = 0; mi < 4; ++mi) {
            union { v16h v; v8h h[2]; } u;
            u.h[0] = *(const v8h*)&sA[p][wm + mi * 16 + l16][half16];
            u.h[1] = *(const v8h*)&sA[p][wm + mi * 16 + l16][half16 + 8];
            const v16h afrag = u.v;
            #pragma unroll
            for (int ni = 0; ni < 4; ++ni)
                acc[mi][ni] = __builtin_amdgcn_wmma_f32_16x16x32_f16(
                    /*neg_a=*/false, afrag,
                    /*neg_b=*/false, bfrag[ni],
                    /*c_mod=*/(short)0, acc[mi][ni],
                    /*reuse_a=*/false, /*reuse_b=*/false);
        }

        if (hasNext)
            commit_stage(p ^ 1);                 // pk-f16 dequant + ds_store next buffer

        __syncthreads();
    }

    // ---- epilogue: C layout (VGPR v, lane) -> M = 8*(lane/16)+v, N = lane%16 ----
    const int mhi = (lane >> 4) * 8;
    #pragma unroll
    for (int mi = 0; mi < 4; ++mi) {
        #pragma unroll
        for (int ni = 0; ni < 4; ++ni) {
            const int col = n0 + wn + ni * 16 + l16;
            const float b = bias[col];
            #pragma unroll
            for (int v = 0; v < 8; ++v) {
                const int row = m0 + wm + mi * 16 + mhi + v;
                out[(size_t)row * N + col] = acc[mi][ni][v] + b;
            }
        }
    }
}

extern "C" void kernel_launch(void* const* d_in, const int* in_sizes, int n_in,
                              void* d_out, int out_size, void* d_ws, size_t ws_size,
                              hipStream_t stream) {
    const float* x       = (const float*)d_in[0];
    const int*   qweight = (const int*)  d_in[1];
    const int*   qzeros  = (const int*)  d_in[2];
    const float* scales  = (const float*)d_in[3];
    const float* bias    = (const float*)d_in[4];
    float*       out     = (float*)d_out;

    const int N = in_sizes[4];                 // outfeatures (bias length)
    const int K = (in_sizes[1] / N) * 8;       // qweight rows * 8 nibbles
    const int M = in_sizes[0] / K;             // tokens

    dim3 grid(N / BLOCK_N, M / BLOCK_M);
    w4a16_gemm_wmma<<<grid, 256, 0, stream>>>(x, qweight, qzeros, scales, bias, out, M, N, K);
}
